// FusedRMSNormQKVRoPE_87505663689136
// MI455X (gfx1250) — matrix operand order repaired and checked by reading, not compiled, not run on hardware
//
#include <hip/hip_runtime.h>
#include <hip/hip_bf16.h>

// ---------------------------------------------------------------------------
// FusedRMSNormQKVRoPE for gfx1250 (MI455X), wave32 + v_wmma_f32_16x16x32_f16
//   phase 1: (x+res)*rsqrt(mean(x^2)+eps) -> f16 A[8192x2048]   (ws)
//   phase 2: W[2048x6144] f32 -> Wt[6144x2048] f16 (transpose)  (ws)
//   phase 3: WMMA GEMM, double-buffered LDS fed by
//            GLOBAL_LOAD_ASYNC_TO_LDS_B128 (ASYNCcnt), + bias + RoPE scatter
// ---------------------------------------------------------------------------

typedef __attribute__((ext_vector_type(16))) _Float16 v16h;
typedef __attribute__((ext_vector_type(8)))  float    v8f;
typedef __attribute__((ext_vector_type(4)))  int      v4i;

union F16Frag { uint4 u[2]; v16h h; };

#define GB 4
#define GS 2048
#define GH 2048
#define GN 6144
#define GM 8192            // B*S
#define QSZ 16777216       // 4*16*2048*128
#define TILE_K 32
#define LDS_PITCH 40       // halves; 80B row pitch -> conflict-free, 16B aligned

#define AS1 __attribute__((address_space(1)))
#define AS3 __attribute__((address_space(3)))

// ---- async global->LDS 16B copy (gfx1250 GLOBAL_LOAD_ASYNC_TO_LDS_B128) ---
__device__ __forceinline__ void async_copy16(const void* g, void* l) {
#if __has_builtin(__builtin_amdgcn_global_load_async_to_lds_b128)
    __builtin_amdgcn_global_load_async_to_lds_b128(
        (AS1 v4i*)(AS1 void*)(void*)g,
        (AS3 v4i*)(AS3 void*)l, 0, 0);
#else
    unsigned lds_off = (unsigned)(unsigned long long)(AS3 void*)l;
    unsigned long long ga = (unsigned long long)g;
    asm volatile("global_load_async_to_lds_b128 %0, %1, off"
                 :: "v"(lds_off), "v"(ga) : "memory");
#endif
}

__device__ __forceinline__ void wait_async0() {
#if __has_builtin(__builtin_amdgcn_s_wait_asynccnt)
    __builtin_amdgcn_s_wait_asynccnt(0);
#else
    asm volatile("s_wait_asynccnt 0x0" ::: "memory");
#endif
}

// ---------------- phase 1: residual add + RMSNorm -> f16 -------------------
__global__ __launch_bounds__(256) void rmsnorm_f16_kernel(
    const float* __restrict__ x, const float* __restrict__ res,
    _Float16* __restrict__ A)
{
    const int m = blockIdx.x;
    const int t = threadIdx.x;
    const float* xr = x   + (size_t)m * GH;
    const float* rr = res + (size_t)m * GH;

    float4 x0 = ((const float4*)xr)[t * 2 + 0];
    float4 x1 = ((const float4*)xr)[t * 2 + 1];
    float ss = x0.x*x0.x + x0.y*x0.y + x0.z*x0.z + x0.w*x0.w
             + x1.x*x1.x + x1.y*x1.y + x1.z*x1.z + x1.w*x1.w;
    #pragma unroll
    for (int o = 16; o > 0; o >>= 1) ss += __shfl_xor(ss, o, 32);

    __shared__ float red[8];
    if ((t & 31) == 0) red[t >> 5] = ss;
    __syncthreads();
    float tot = 0.f;
    #pragma unroll
    for (int w = 0; w < 8; ++w) tot += red[w];
    const float scale = rsqrtf(tot * (1.0f / (float)GH) + 1e-6f);

    float4 r0 = ((const float4*)rr)[t * 2 + 0];
    float4 r1 = ((const float4*)rr)[t * 2 + 1];
    union { _Float16 h[8]; uint4 u; } o;
    o.h[0] = (_Float16)((x0.x + r0.x) * scale);
    o.h[1] = (_Float16)((x0.y + r0.y) * scale);
    o.h[2] = (_Float16)((x0.z + r0.z) * scale);
    o.h[3] = (_Float16)((x0.w + r0.w) * scale);
    o.h[4] = (_Float16)((x1.x + r1.x) * scale);
    o.h[5] = (_Float16)((x1.y + r1.y) * scale);
    o.h[6] = (_Float16)((x1.z + r1.z) * scale);
    o.h[7] = (_Float16)((x1.w + r1.w) * scale);
    ((uint4*)A)[(size_t)m * (GH / 8) + t] = o.u;
}

// ---------------- phase 2: W (KxN f32) -> Wt (NxK f16) ---------------------
__global__ __launch_bounds__(256) void wt_transpose_kernel(
    const float* __restrict__ W, _Float16* __restrict__ Wt)
{
    __shared__ float tile[32][33];
    const int tx = threadIdx.x & 31;
    const int ty = threadIdx.x >> 5;           // 0..7
    const int n0 = blockIdx.x * 32;
    const int k0 = blockIdx.y * 32;
    #pragma unroll
    for (int r = 0; r < 4; ++r) {
        int k = k0 + ty + r * 8;
        tile[ty + r * 8][tx] = W[(size_t)k * GN + n0 + tx];
    }
    __syncthreads();
    #pragma unroll
    for (int r = 0; r < 4; ++r) {
        int n = n0 + ty + r * 8;
        Wt[(size_t)n * GH + k0 + tx] = (_Float16)tile[tx][ty + r * 8];
    }
}

// ---------------- phase 3: WMMA GEMM + bias + RoPE + scatter ---------------
__global__ __launch_bounds__(256) void qkv_gemm_rope_kernel(
    const _Float16* __restrict__ A,    // [8192][2048]
    const _Float16* __restrict__ Wt,   // [6144][2048]
    const float* __restrict__ bias,    // [6144]
    const float* __restrict__ cosT,    // [2048][64]
    const float* __restrict__ sinT,    // [2048][64]
    float* __restrict__ out)           // q | k | v, each (4,16,2048,128)
{
    __shared__ _Float16 sA[2][128 * LDS_PITCH];
    __shared__ _Float16 sB[2][128 * LDS_PITCH];

    const int tid  = threadIdx.x;
    const int lane = tid & 31;
    const int wave = tid >> 5;
    const int wm = wave >> 2;          // 0..1 : 64-row strip
    const int wn = wave & 3;           // 0..3 : 32-col strip
    const int m0 = blockIdx.y * 128;
    const int n0 = blockIdx.x * 128;

    // staging: thread t fills rows r0 and r0+64, 16B sub-chunk sub0
    const int r0   = tid >> 2;
    const int sub0 = tid & 3;
    const _Float16* gA0 = A  + (size_t)(m0 + r0) * GH + sub0 * 8;
    const _Float16* gA1 = gA0 + (size_t)64 * GH;
    const _Float16* gB0 = Wt + (size_t)(n0 + r0) * GH + sub0 * 8;
    const _Float16* gB1 = gB0 + (size_t)64 * GH;
    const int lw0 = r0 * LDS_PITCH + sub0 * 8;
    const int lw1 = (r0 + 64) * LDS_PITCH + sub0 * 8;

    // fragment LDS byte offsets (ISA 7.12.2 layouts)
    const int col = lane & 15;
    const int hi  = lane >> 4;
    const int offA0 = hi * 16;          // A: K blocks {0..7,16..23}/{8..15,24..31}
    const int offA1 = offA0 + 32;
    const int offB0 = hi * 16;          // B: K blocks {0..7,16..23}/{8..15,24..31}
    const int offB1 = offB0 + 32;

    v8f acc[4][2] = {};

    // prologue: async-stage k-chunk 0 into buffer 0
    async_copy16(gA0, &sA[0][lw0]);
    async_copy16(gA1, &sA[0][lw1]);
    async_copy16(gB0, &sB[0][lw0]);
    async_copy16(gB1, &sB[0][lw1]);

    int cur = 0;
    const int KSTEPS = GH / TILE_K;    // 64
    for (int kt = 0; kt < KSTEPS; ++kt) {
        wait_async0();                 // my loads into buf[cur] have landed
        __syncthreads();               // everyone's landed; prev reads finished

        if (kt + 1 < KSTEPS) {         // overlap next-tile DMA with compute
            const int ko = (kt + 1) * TILE_K;
            const int nb = cur ^ 1;
            async_copy16(gA0 + ko, &sA[nb][lw0]);
            async_copy16(gA1 + ko, &sA[nb][lw1]);
            async_copy16(gB0 + ko, &sB[nb][lw0]);
            async_copy16(gB1 + ko, &sB[nb][lw1]);
        }

        F16Frag af[4];
        #pragma unroll
        for (int i = 0; i < 4; ++i) {
            const char* base = (const char*)&sA[cur][0]
                             + (size_t)(wm * 64 + i * 16 + col) * (LDS_PITCH * 2);
            af[i].u[0] = *(const uint4*)(base + offA0);
            af[i].u[1] = *(const uint4*)(base + offA1);
        }
        F16Frag bf[2];
        #pragma unroll
        for (int j = 0; j < 2; ++j) {
            const char* base = (const char*)&sB[cur][0]
                             + (size_t)(wn * 32 + j * 16 + col) * (LDS_PITCH * 2);
            bf[j].u[0] = *(const uint4*)(base + offB0);
            bf[j].u[1] = *(const uint4*)(base + offB1);
        }
        #pragma unroll
        for (int i = 0; i < 4; ++i)
            #pragma unroll
            for (int j = 0; j < 2; ++j)
                acc[i][j] = __builtin_amdgcn_wmma_f32_16x16x32_f16(
                    false, af[i].h, false, bf[j].h,
                    (short)0, acc[i][j], false, false);

        cur ^= 1;
    }

    // epilogue: bias + RoPE + scatter.  Whole block is one (segment, head).
    const int seg  = blockIdx.x >> 4;   // 0=q, 1=k, 2=v
    const int head = blockIdx.x & 15;
    float* outSeg = out + (size_t)seg * QSZ;

    #pragma unroll
    for (int j = 0; j < 2; ++j) {
        const int n = n0 + wn * 32 + j * 16 + col;
        const int d = n & 127;
        const float bval = bias[n];
        #pragma unroll
        for (int i = 0; i < 4; ++i) {
            v8f a = acc[i][j];
            #pragma unroll
            for (int g = 0; g < 8; ++g) {
                const int m = m0 + wm * 64 + i * 16 + hi * 8 + g;
                const int b = m >> 11;           // m / 2048
                const int s = m & 2047;
                float v = a[g] + bval;
                float outv;
                if (seg < 2) {
                    const float cc = cosT[s * 64 + (d >> 1)];
                    const float sn = sinT[s * 64 + (d >> 1)];
                    const float partner = __shfl_xor(v, 1, 32);
                    outv = (d & 1) ? (partner * sn + v * cc)
                                   : (v * cc - partner * sn);
                } else {
                    outv = v;
                }
                outSeg[((size_t)(b * 16 + head) * GS + s) * 128 + d] = outv;
            }
        }
    }
}

// ---------------------------------------------------------------------------
extern "C" void kernel_launch(void* const* d_in, const int* in_sizes, int n_in,
                              void* d_out, int out_size, void* d_ws, size_t ws_size,
                              hipStream_t stream) {
    const float* x        = (const float*)d_in[0];
    const float* residual = (const float*)d_in[1];
    const float* weight   = (const float*)d_in[2];
    const float* bias     = (const float*)d_in[3];
    const float* cosT     = (const float*)d_in[4];
    const float* sinT     = (const float*)d_in[5];
    float* out = (float*)d_out;

    _Float16* Af = (_Float16*)d_ws;                                // 33,554,432 B
    _Float16* Wt = (_Float16*)((char*)d_ws + (size_t)GM * GH * 2); // 25,165,824 B

    rmsnorm_f16_kernel<<<GM, 256, 0, stream>>>(x, residual, Af);
    wt_transpose_kernel<<<dim3(GN / 32, GH / 32), 256, 0, stream>>>(weight, Wt);
    qkv_gemm_rope_kernel<<<dim3(GN / 128, GM / 128), 256, 0, stream>>>(
        Af, Wt, bias, cosT, sinT, out);
}
